// GATTransformer_47596827574855
// MI455X (gfx1250) — matrix operand order repaired and checked by reading, not compiled, not run on hardware
//
#include <hip/hip_runtime.h>
#include <hip/hip_bf16.h>
#include <math.h>

// ---------------- problem constants ----------------
#define NN    20000
#define EE    320000
#define DIMD  128
#define HH    2
#define HCC   256      // H*DIM
#define MLPD  512
#define DEPTHD 2
#define SLOPE 0.2f

typedef __attribute__((ext_vector_type(16))) _Float16 v16h;
typedef __attribute__((ext_vector_type(8)))  _Float16 v8h;
typedef __attribute__((ext_vector_type(8)))  float    v8f;

// ---------------- helpers ----------------
__device__ __forceinline__ unsigned ord_of_float(float f) {
    unsigned u = __float_as_uint(f);
    return (u & 0x80000000u) ? ~u : (u | 0x80000000u);
}
__device__ __forceinline__ float float_of_ord(unsigned o) {
    unsigned u = (o & 0x80000000u) ? (o & 0x7FFFFFFFu) : ~o;
    return __uint_as_float(u);
}

// ---------------- elementwise utility kernels ----------------
__global__ void copy_f32_kernel(const float* __restrict__ in, float* __restrict__ out, int n) {
    int i = blockIdx.x * blockDim.x + threadIdx.x;
    if (i < n) out[i] = in[i];
}

__global__ void f32_to_f16_kernel(const float* __restrict__ in, _Float16* __restrict__ out, int n) {
    int i = blockIdx.x * blockDim.x + threadIdx.x;
    if (i < n) out[i] = (_Float16)in[i];
}

// W: [K, Nc] f32 row-major  ->  Wt: [Nc, K] f16 row-major (K contiguous per column)
__global__ void wtrans_kernel(const float* __restrict__ W, _Float16* __restrict__ Wt, int K, int Nc) {
    int i = blockIdx.x * blockDim.x + threadIdx.x;
    if (i >= K * Nc) return;
    int k = i / Nc, n = i % Nc;
    Wt[(size_t)n * K + k] = (_Float16)W[i];
}

// agg[n, j] = bias_g[j];  mx = 0; zsum = 0
__global__ void init_seg_kernel(float* __restrict__ agg, const float* __restrict__ bias_g,
                                unsigned* __restrict__ mx, float* __restrict__ zsum) {
    int i = blockIdx.x * blockDim.x + threadIdx.x;
    if (i < NN * HCC) agg[i] = bias_g[i & (HCC - 1)];
    if (i < NN * HH) { mx[i] = 0u; zsum[i] = 0.0f; }
}

// ---------------- LayerNorm: one wave per row of 128, f16 output ----------------
__global__ __launch_bounds__(256)
void ln_kernel(const float* __restrict__ x, const float* __restrict__ w,
               const float* __restrict__ b, _Float16* __restrict__ out) {
    int wave = threadIdx.x >> 5;
    int lane = threadIdx.x & 31;
    int row  = blockIdx.x * 8 + wave;
    if (row >= NN) return;
    const float4* xp = (const float4*)(x + (size_t)row * DIMD);
    float4 v = xp[lane];
    float s  = v.x + v.y + v.z + v.w;
    float s2 = v.x * v.x + v.y * v.y + v.z * v.z + v.w * v.w;
    #pragma unroll
    for (int off = 16; off > 0; off >>= 1) {
        s  += __shfl_xor(s,  off, 32);
        s2 += __shfl_xor(s2, off, 32);
    }
    float mean = s * (1.0f / DIMD);
    float var  = s2 * (1.0f / DIMD) - mean * mean;
    float rstd = rsqrtf(var + 1e-5f);
    float4 wv = ((const float4*)w)[lane];
    float4 bv = ((const float4*)b)[lane];
    _Float16* op = out + (size_t)row * DIMD + lane * 4;
    op[0] = (_Float16)((v.x - mean) * rstd * wv.x + bv.x);
    op[1] = (_Float16)((v.y - mean) * rstd * wv.y + bv.y);
    op[2] = (_Float16)((v.z - mean) * rstd * wv.z + bv.z);
    op[3] = (_Float16)((v.w - mean) * rstd * wv.w + bv.w);
}

// ---------------- WMMA GEMM: C[M,Nc] = A[M,K](f16) @ Bt[Nc,K]^T(f16) + bias ----------------
// act: 0 -> Cf = acc + bias
//      1 -> Cf = acc + bias + resid   (resid may alias Cf; each thread owns its element)
//      2 -> Ch = gelu(acc + bias)     (exact gelu, f16 output)
__global__ __launch_bounds__(256)
void gemm_wmma_kernel(const _Float16* __restrict__ A, const _Float16* __restrict__ Bt,
                      const float* __restrict__ bias, const float* __restrict__ resid,
                      float* __restrict__ Cf, _Float16* __restrict__ Ch,
                      int K, int Nc, int act) {
    int wave = threadIdx.x >> 5;
    int lane = threadIdx.x & 31;
    int m0 = blockIdx.x * 16;
    int n0 = (blockIdx.y * 8 + wave) * 16;
    int hl = lane & 15;        // row (A) / col (B,C) within tile
    int hi = lane >> 4;        // half-wave selector

    const _Float16* ap = A  + (size_t)(m0 + hl) * K + hi * 8;   // A: lane holds row m0+hl
    const _Float16* bp = Bt + (size_t)(n0 + hl) * K + hi * 8;   // B: lane holds col n0+hl

    v8f acc = {};
    for (int ko = 0; ko < K; ko += 32) {
        // A fragment (16x32 f16): lanes 0-15 get K ko+0..7 / ko+16..23; lanes 16-31 get +8 / +24
        v8h alo = *(const v8h*)(ap + ko);
        v8h ahi = *(const v8h*)(ap + ko + 16);
        v16h a;
        #pragma unroll
        for (int i = 0; i < 8; ++i) { a[i] = alo[i]; a[i + 8] = ahi[i]; }
        // B fragment (32x16 f16): lanes 0-15 get K ko+0..7 / ko+16..23; lanes 16-31 get +8 / +24
        v8h blo = *(const v8h*)(bp + ko);
        v8h bhi = *(const v8h*)(bp + ko + 16);
        v16h bmat;
        #pragma unroll
        for (int i = 0; i < 8; ++i) { bmat[i] = blo[i]; bmat[i + 8] = bhi[i]; }
        acc = __builtin_amdgcn_wmma_f32_16x16x32_f16(false, a, false, bmat,
                                                     (short)0, acc, false, false);
    }

    int col  = n0 + hl;
    int row0 = m0 + hi * 8;          // C layout: VGPR v -> row m0 + hi*8 + v, col = n0 + (lane&15)
    float bv = bias[col];

    if (act == 2) {
        #pragma unroll
        for (int v = 0; v < 8; ++v) {
            float t = acc[v] + bv;
            float g = 0.5f * t * (1.0f + erff(t * 0.70710678118654752f));
            Ch[(size_t)(row0 + v) * Nc + col] = (_Float16)g;
        }
    } else if (act == 1) {
        #pragma unroll
        for (int v = 0; v < 8; ++v) {
            size_t idx = (size_t)(row0 + v) * Nc + col;
            Cf[idx] = acc[v] + bv + resid[idx];
        }
    } else {
        #pragma unroll
        for (int v = 0; v < 8; ++v) {
            size_t idx = (size_t)(row0 + v) * Nc + col;
            Cf[idx] = acc[v] + bv;
        }
    }
}

// ---------------- edge pass 1: score + segment max (ordered-uint atomic max) ----------------
__global__ void edge_score_kernel(const int* __restrict__ src, const int* __restrict__ dst,
                                  const float* __restrict__ gl, const float* __restrict__ gr,
                                  const float* __restrict__ att,   // [H*DIM] for this depth
                                  float* __restrict__ score, unsigned* __restrict__ mx) {
    int t = blockIdx.x * blockDim.x + threadIdx.x;
    if (t >= EE * HH) return;
    int e = t >> 1, h = t & 1;
    int s = src[e], d = dst[e];
    const float4* glp = (const float4*)(gl + (size_t)s * HCC + h * DIMD);
    const float4* grp = (const float4*)(gr + (size_t)d * HCC + h * DIMD);
    const float4* ap  = (const float4*)(att + h * DIMD);
    float acc = 0.0f;
    #pragma unroll 4
    for (int i = 0; i < DIMD / 4; ++i) {
        float4 a = glp[i], b = grp[i], w = ap[i];
        float zx = a.x + b.x; zx = zx > 0.0f ? zx : SLOPE * zx; acc += zx * w.x;
        float zy = a.y + b.y; zy = zy > 0.0f ? zy : SLOPE * zy; acc += zy * w.y;
        float zz = a.z + b.z; zz = zz > 0.0f ? zz : SLOPE * zz; acc += zz * w.z;
        float zw = a.w + b.w; zw = zw > 0.0f ? zw : SLOPE * zw; acc += zw * w.w;
    }
    score[t] = acc;
    atomicMax(&mx[d * HH + h], ord_of_float(acc));
}

// ---------------- edge pass 2: exp(score - max) + segment sum ----------------
__global__ void edge_exp_kernel(const int* __restrict__ dst, const float* __restrict__ score,
                                const unsigned* __restrict__ mx, float* __restrict__ expsc,
                                float* __restrict__ zsum) {
    int t = blockIdx.x * blockDim.x + threadIdx.x;
    if (t >= EE * HH) return;
    int e = t >> 1, h = t & 1;
    int d = dst[e];
    unsigned mo = mx[d * HH + h];
    float m = (mo == 0u) ? 0.0f : float_of_ord(mo);   // empty segment -> 0 (matches reference)
    float ex = expf(score[t] - m);
    expsc[t] = ex;
    atomicAdd(&zsum[d * HH + h], ex);
}

// ---------------- edge pass 3: agg[dst] += gl[src] * alpha ----------------
__global__ void edge_agg_kernel(const int* __restrict__ src, const int* __restrict__ dst,
                                const float* __restrict__ gl, const float* __restrict__ expsc,
                                const float* __restrict__ zsum, float* __restrict__ agg) {
    int t = blockIdx.x * blockDim.x + threadIdx.x;   // E * 64 threads, float4 each
    if (t >= EE * 64) return;
    int e = t >> 6, q = t & 63;
    int h = q >> 5;                                   // first 32 float4s = head 0
    int s = src[e], d = dst[e];
    float alpha = expsc[e * HH + h] / (zsum[d * HH + h] + 1e-16f);
    float4 g = ((const float4*)(gl + (size_t)s * HCC))[q];
    float* out = agg + (size_t)d * HCC + q * 4;
    atomicAdd(out + 0, g.x * alpha);
    atomicAdd(out + 1, g.y * alpha);
    atomicAdd(out + 2, g.z * alpha);
    atomicAdd(out + 3, g.w * alpha);
}

// ---------------- host launcher ----------------
extern "C" void kernel_launch(void* const* d_in, const int* in_sizes, int n_in,
                              void* d_out, int out_size, void* d_ws, size_t ws_size,
                              hipStream_t stream) {
    const float* x_in   = (const float*)d_in[0];
    const int*   ei     = (const int*)  d_in[1];
    const float* ln1_w  = (const float*)d_in[2];
    const float* ln1_b  = (const float*)d_in[3];
    const float* Wl     = (const float*)d_in[4];
    const float* bl     = (const float*)d_in[5];
    const float* Wr     = (const float*)d_in[6];
    const float* br     = (const float*)d_in[7];
    const float* att    = (const float*)d_in[8];
    const float* bias_g = (const float*)d_in[9];
    const float* Wq     = (const float*)d_in[10];
    const float* bq     = (const float*)d_in[11];
    const float* ln2_w  = (const float*)d_in[12];
    const float* ln2_b  = (const float*)d_in[13];
    const float* W1     = (const float*)d_in[14];
    const float* b1     = (const float*)d_in[15];
    const float* W2     = (const float*)d_in[16];
    const float* b2     = (const float*)d_in[17];

    const int* src = ei;
    const int* dst = ei + EE;

    // -------- workspace carving --------
    char* p = (char*)d_ws;
    auto alloc = [&](size_t bytes) -> char* {
        char* r = p; p += (bytes + 255) & ~(size_t)255; return r;
    };
    float*    xbuf   = (float*)   alloc((size_t)NN * DIMD * 4);
    _Float16* hF16   = (_Float16*)alloc((size_t)NN * DIMD * 2);
    float*    gl     = (float*)   alloc((size_t)NN * HCC * 4);
    float*    gr     = (float*)   alloc((size_t)NN * HCC * 4);
    float*    agg    = (float*)   alloc((size_t)NN * HCC * 4);
    _Float16* aggF16 = (_Float16*)alloc((size_t)NN * HCC * 2);
    _Float16* midF16 = (_Float16*)alloc((size_t)NN * MLPD * 2);
    float*    score  = (float*)   alloc((size_t)EE * HH * 4);
    float*    expsc  = (float*)   alloc((size_t)EE * HH * 4);
    unsigned* mx     = (unsigned*)alloc((size_t)NN * HH * 4);
    float*    zsum   = (float*)   alloc((size_t)NN * HH * 4);
    _Float16* WlT[DEPTHD]; _Float16* WrT[DEPTHD]; _Float16* WqT[DEPTHD];
    _Float16* W1T[DEPTHD]; _Float16* W2T[DEPTHD];
    for (int d = 0; d < DEPTHD; ++d) {
        WlT[d] = (_Float16*)alloc((size_t)DIMD * HCC * 2);
        WrT[d] = (_Float16*)alloc((size_t)DIMD * HCC * 2);
        WqT[d] = (_Float16*)alloc((size_t)HCC * DIMD * 2);
        W1T[d] = (_Float16*)alloc((size_t)DIMD * MLPD * 2);
        W2T[d] = (_Float16*)alloc((size_t)MLPD * DIMD * 2);
    }

    const int TB = 256;
    auto blks = [](long n, int tb) { return (int)((n + tb - 1) / tb); };

    // x working copy
    copy_f32_kernel<<<blks((long)NN * DIMD, TB), TB, 0, stream>>>(x_in, xbuf, NN * DIMD);

    // transpose+convert all weights to f16 [Nc, K] once
    for (int d = 0; d < DEPTHD; ++d) {
        wtrans_kernel<<<blks((long)DIMD * HCC, TB), TB, 0, stream>>>(Wl + (size_t)d * DIMD * HCC, WlT[d], DIMD, HCC);
        wtrans_kernel<<<blks((long)DIMD * HCC, TB), TB, 0, stream>>>(Wr + (size_t)d * DIMD * HCC, WrT[d], DIMD, HCC);
        wtrans_kernel<<<blks((long)HCC * DIMD, TB), TB, 0, stream>>>(Wq + (size_t)d * HCC * DIMD, WqT[d], HCC, DIMD);
        wtrans_kernel<<<blks((long)DIMD * MLPD, TB), TB, 0, stream>>>(W1 + (size_t)d * DIMD * MLPD, W1T[d], DIMD, MLPD);
        wtrans_kernel<<<blks((long)MLPD * DIMD, TB), TB, 0, stream>>>(W2 + (size_t)d * MLPD * DIMD, W2T[d], MLPD, DIMD);
    }

    for (int d = 0; d < DEPTHD; ++d) {
        // ---- PreNorm 1 ----
        ln_kernel<<<NN / 8, TB, 0, stream>>>(xbuf, ln1_w + (size_t)d * DIMD, ln1_b + (size_t)d * DIMD, hF16);

        // ---- gl = h@Wl + bl ; gr = h@Wr + br   (WMMA) ----
        dim3 g1(NN / 16, HCC / 128);
        gemm_wmma_kernel<<<g1, TB, 0, stream>>>(hF16, WlT[d], bl + (size_t)d * HCC, nullptr, gl, nullptr, DIMD, HCC, 0);
        gemm_wmma_kernel<<<g1, TB, 0, stream>>>(hF16, WrT[d], br + (size_t)d * HCC, nullptr, gr, nullptr, DIMD, HCC, 0);

        // ---- segment softmax attention ----
        init_seg_kernel<<<blks((long)NN * HCC, TB), TB, 0, stream>>>(agg, bias_g + (size_t)d * HCC, mx, zsum);
        edge_score_kernel<<<blks((long)EE * HH, TB), TB, 0, stream>>>(src, dst, gl, gr, att + (size_t)d * HH * DIMD, score, mx);
        edge_exp_kernel<<<blks((long)EE * HH, TB), TB, 0, stream>>>(dst, score, mx, expsc, zsum);
        edge_agg_kernel<<<blks((long)EE * 64, TB), TB, 0, stream>>>(src, dst, gl, expsc, zsum, agg);

        // ---- x = (agg + bias_g)@Wq + bq + x   (WMMA) ----
        f32_to_f16_kernel<<<blks((long)NN * HCC, TB), TB, 0, stream>>>(agg, aggF16, NN * HCC);
        dim3 g2(NN / 16, DIMD / 128);
        gemm_wmma_kernel<<<g2, TB, 0, stream>>>(aggF16, WqT[d], bq + (size_t)d * DIMD, xbuf, xbuf, nullptr, HCC, DIMD, 1);

        // ---- PreNorm 2 + MLP ----
        ln_kernel<<<NN / 8, TB, 0, stream>>>(xbuf, ln2_w + (size_t)d * DIMD, ln2_b + (size_t)d * DIMD, hF16);
        dim3 g3(NN / 16, MLPD / 128);
        gemm_wmma_kernel<<<g3, TB, 0, stream>>>(hF16, W1T[d], b1 + (size_t)d * MLPD, nullptr, nullptr, midF16, DIMD, MLPD, 2);
        dim3 g4(NN / 16, DIMD / 128);
        gemm_wmma_kernel<<<g4, TB, 0, stream>>>(midF16, W2T[d], b2 + (size_t)d * DIMD, xbuf, xbuf, nullptr, MLPD, DIMD, 1);
    }

    copy_f32_kernel<<<blks((long)NN * DIMD, TB), TB, 0, stream>>>(xbuf, (float*)d_out, NN * DIMD);
}
